// MHSA_20822001451106
// MI455X (gfx1250) — matrix operand order repaired and checked by reading, not compiled, not run on hardware
//
#include <hip/hip_runtime.h>

// ---------------------------------------------------------------------------
// MHSA (channel-attention variant) for MI455X / gfx1250.
// All GEMMs on v_wmma_f32_16x16x32_bf16 (bf16 in, f32 accumulate).
// LDS double-buffered pipeline: async global->LDS prefetch of tile k+1
// (ASYNCcnt path) overlapped with WMMA compute on tile k.
// ---------------------------------------------------------------------------

typedef __attribute__((ext_vector_type(16))) __bf16 v16bf;
typedef __attribute__((ext_vector_type(8)))  __bf16 v8bf;
typedef __attribute__((ext_vector_type(4)))  __bf16 v4bf;
typedef __attribute__((ext_vector_type(8)))  float  v8f;

#define BATCH 8
#define CDIM  512
#define NDIM  4096      // H*W
#define ODIM  1536      // 3*C
#define HEADS 8
#define DHEAD 64

static __device__ __forceinline__ v16bf cat8(v8bf lo, v8bf hi) {
  return __builtin_shufflevector(lo, hi, 0,1,2,3,4,5,6,7,8,9,10,11,12,13,14,15);
}

static __device__ __forceinline__ v8f wmma_bf16(v16bf a, v16bf b, v8f c) {
  // (neg_a, A, neg_b, B, c_mod, C, reuse_a, reuse_b)
  return __builtin_amdgcn_wmma_f32_16x16x32_bf16(false, a, false, b, (short)0, c,
                                                 false, false);
}

static __device__ __forceinline__ v8f vzero8() {
  v8f z = {0.f,0.f,0.f,0.f,0.f,0.f,0.f,0.f};
  return z;
}

// Async 16-byte global -> LDS copy (per active lane). Low 32 bits of a flat
// LDS address are the wave-relative LDS byte offset (ISA aperture rules).
static __device__ __forceinline__ void async_g2l_b128(void* lds_ptr,
                                                      const void* gptr) {
  unsigned lds_off = (unsigned)(size_t)lds_ptr;
  unsigned long long ga = (unsigned long long)(size_t)gptr;
  asm volatile("global_load_async_to_lds_b128 %0, %1, off"
               :: "v"(lds_off), "v"(ga) : "memory");
}
static __device__ __forceinline__ void wait_async0() {
  asm volatile("s_wait_asynccnt 0x0" ::: "memory");
}

// pack two values into one 32-bit LDS word (adjacent k in [n][k] layout)
static __device__ __forceinline__ unsigned pack_bf16_f32(float lo, float hi) {
  unsigned short a = __builtin_bit_cast(unsigned short, (__bf16)lo);
  unsigned short b = __builtin_bit_cast(unsigned short, (__bf16)hi);
  return (unsigned)a | ((unsigned)b << 16);
}
static __device__ __forceinline__ unsigned pack_bf16(__bf16 lo, __bf16 hi) {
  unsigned short a = __builtin_bit_cast(unsigned short, lo);
  unsigned short b = __builtin_bit_cast(unsigned short, hi);
  return (unsigned)a | ((unsigned)b << 16);
}

// --------------------------- weight conversion -----------------------------
__global__ void f32_to_bf16_kernel(const float* __restrict__ s,
                                   __bf16* __restrict__ d, int n) {
  int i = blockIdx.x * blockDim.x + threadIdx.x;
  int stride = gridDim.x * blockDim.x;
  for (; i < n; i += stride) d[i] = (__bf16)s[i];
}

// ------------------------------ GEMM 1: QKV --------------------------------
// Out[b][o][n] = bf16( sum_c Wbf[o][c] * X[b][c][n] + bias[o] )
// Block: 256 thr, tile 128(M) x 64(N); each wave 32x32 (2x2 wmma tiles).
__global__ void qkv_gemm_kernel(const __bf16* __restrict__ Wbf,
                                const float*  __restrict__ X,
                                const float*  __restrict__ bias,
                                __bf16* __restrict__ Out) {
  __shared__ __bf16 As[2][128][32];   // [m][k]  2 x 8KB
  __shared__ __bf16 Bst[2][64][32];   // [n][k]  2 x 4KB (transposed)

  const int b  = blockIdx.z;
  const int m0 = blockIdx.y * 128;
  const int n0 = blockIdx.x * 64;
  const float* Xb = X + (size_t)b * CDIM * NDIM;

  const int t    = threadIdx.x;
  const int lane = t & 31;
  const int w    = t >> 5;
  const int lrow = lane & 15;
  const int lhi  = lane >> 4;
  const int mw   = w & 3;          // wave row group (32 rows)
  const int nw   = w >> 2;         // wave col group (32 cols)

  // staging maps
  const int ar0s = (t)       >> 2, ac0s = ((t)       & 3) * 8;  // A copy 0
  const int ar1s = (t + 256) >> 2, ac1s = ((t + 256) & 3) * 8;  // A copy 1
  const int nb  = (t & 15) * 4;    // B: 4 n-values
  const int rk0 = (t >> 4) * 2;    // B: k row pair
  const int rk1 = rk0 + 1;

  v8f acc00 = vzero8(), acc01 = vzero8(), acc10 = vzero8(), acc11 = vzero8();

  // ---- prologue: stage tile 0 into buffer 0 ----
  async_g2l_b128(&As[0][ar0s][ac0s], Wbf + (size_t)(m0 + ar0s) * CDIM + ac0s);
  async_g2l_b128(&As[0][ar1s][ac1s], Wbf + (size_t)(m0 + ar1s) * CDIM + ac1s);
  {
    float4 g0 = *(const float4*)(Xb + (size_t)rk0 * NDIM + n0 + nb);
    float4 g1 = *(const float4*)(Xb + (size_t)rk1 * NDIM + n0 + nb);
    *(unsigned*)(&Bst[0][nb + 0][rk0]) = pack_bf16_f32(g0.x, g1.x);
    *(unsigned*)(&Bst[0][nb + 1][rk0]) = pack_bf16_f32(g0.y, g1.y);
    *(unsigned*)(&Bst[0][nb + 2][rk0]) = pack_bf16_f32(g0.z, g1.z);
    *(unsigned*)(&Bst[0][nb + 3][rk0]) = pack_bf16_f32(g0.w, g1.w);
  }
  wait_async0();
  __syncthreads();

  const int KT = CDIM / 32;
  for (int kt = 0; kt < KT; ++kt) {
    const int cur = kt & 1, nxt = cur ^ 1;
    const bool has_next = (kt + 1) < KT;
    float4 g0, g1;
    if (has_next) {  // prefetch tile kt+1 while computing tile kt
      const int k0n = (kt + 1) * 32;
      async_g2l_b128(&As[nxt][ar0s][ac0s],
                     Wbf + (size_t)(m0 + ar0s) * CDIM + k0n + ac0s);
      async_g2l_b128(&As[nxt][ar1s][ac1s],
                     Wbf + (size_t)(m0 + ar1s) * CDIM + k0n + ac1s);
      g0 = *(const float4*)(Xb + (size_t)(k0n + rk0) * NDIM + n0 + nb);
      g1 = *(const float4*)(Xb + (size_t)(k0n + rk1) * NDIM + n0 + nb);
    }

    const int ar = mw * 32 + lrow;
    v16bf a0 = *(const v16bf*)(&As[cur][ar][lhi * 16]);
    v16bf a1 = *(const v16bf*)(&As[cur][ar + 16][lhi * 16]);
    v16bf b0 = *(const v16bf*)(&Bst[cur][nw * 32 + lrow][lhi * 16]);
    v16bf b1 = *(const v16bf*)(&Bst[cur][nw * 32 + 16 + lrow][lhi * 16]);
    acc00 = wmma_bf16(a0, b0, acc00);
    acc01 = wmma_bf16(a0, b1, acc01);
    acc10 = wmma_bf16(a1, b0, acc10);
    acc11 = wmma_bf16(a1, b1, acc11);

    if (has_next) {
      *(unsigned*)(&Bst[nxt][nb + 0][rk0]) = pack_bf16_f32(g0.x, g1.x);
      *(unsigned*)(&Bst[nxt][nb + 1][rk0]) = pack_bf16_f32(g0.y, g1.y);
      *(unsigned*)(&Bst[nxt][nb + 2][rk0]) = pack_bf16_f32(g0.z, g1.z);
      *(unsigned*)(&Bst[nxt][nb + 3][rk0]) = pack_bf16_f32(g0.w, g1.w);
    }
    wait_async0();
    __syncthreads();
  }

  const int nA = n0 + nw * 32 + lrow;
  const int nB = nA + 16;
#pragma unroll
  for (int i = 0; i < 8; ++i) {
    const int o0 = m0 + mw * 32 + i + 8 * lhi;
    const int o1 = o0 + 16;
    const float bo0 = bias[o0], bo1 = bias[o1];
    Out[((size_t)b * ODIM + o0) * NDIM + nA] = (__bf16)(acc00[i] + bo0);
    Out[((size_t)b * ODIM + o0) * NDIM + nB] = (__bf16)(acc01[i] + bo0);
    Out[((size_t)b * ODIM + o1) * NDIM + nA] = (__bf16)(acc10[i] + bo1);
    Out[((size_t)b * ODIM + o1) * NDIM + nB] = (__bf16)(acc11[i] + bo1);
  }
}

// --------------------- Attention logits + softmax --------------------------
// One block per (b,h). logits[d][e] = scale * sum_n q[d][n]*k[e][n] (K=4096)
// softmax over e; writes attn (bf16, 64x64 per head) to workspace.
__global__ void attn_logits_kernel(const __bf16* __restrict__ qkvbf,
                                   __bf16* __restrict__ attnw) {
  __shared__ __bf16 Qs[2][64][32];   // 2 x 4KB
  __shared__ __bf16 Ks[2][64][32];   // 2 x 4KB
  __shared__ float  L[64][64];       // 16KB

  const int bh = blockIdx.x;
  const int b  = bh >> 3;
  const int h  = bh & 7;
  const __bf16* qp = qkvbf + ((size_t)b * ODIM + h * DHEAD) * NDIM;
  const __bf16* kp = qp + (size_t)CDIM * NDIM;

  const int t    = threadIdx.x;
  const int lane = t & 31;
  const int w    = t >> 5;
  const int lrow = lane & 15;
  const int lhi  = lane >> 4;
  const int mt   = w & 3;
  const int et0  = (w >> 2) * 2;
  const int et1  = et0 + 1;

  const int rs = t >> 2, cs = (t & 3) * 8;   // staging map

  v8f acc0 = vzero8();
  v8f acc1 = vzero8();

  // prologue: stage tile 0
  async_g2l_b128(&Qs[0][rs][cs], qp + (size_t)rs * NDIM + cs);
  async_g2l_b128(&Ks[0][rs][cs], kp + (size_t)rs * NDIM + cs);
  wait_async0();
  __syncthreads();

  const int KT = NDIM / 32;
  for (int kt = 0; kt < KT; ++kt) {
    const int cur = kt & 1, nxt = cur ^ 1;
    if (kt + 1 < KT) {
      const int k0n = (kt + 1) * 32;
      async_g2l_b128(&Qs[nxt][rs][cs], qp + (size_t)rs * NDIM + k0n + cs);
      async_g2l_b128(&Ks[nxt][rs][cs], kp + (size_t)rs * NDIM + k0n + cs);
    }

    const int arow = mt * 16 + lrow;
    v16bf a = *(const v16bf*)(&Qs[cur][arow][lhi * 16]);
    v16bf b0 = *(const v16bf*)(&Ks[cur][et0 * 16 + lrow][lhi * 16]);
    v16bf b1 = *(const v16bf*)(&Ks[cur][et1 * 16 + lrow][lhi * 16]);
    acc0 = wmma_bf16(a, b0, acc0);
    acc1 = wmma_bf16(a, b1, acc1);

    wait_async0();
    __syncthreads();
  }

#pragma unroll
  for (int i = 0; i < 8; ++i) {
    const int r = mt * 16 + i + 8 * lhi;
    L[r][et0 * 16 + lrow] = acc0[i];
    L[r][et1 * 16 + lrow] = acc1[i];
  }
  __syncthreads();

  if (t < 64) {
    const float scale = 0.125f;    // 1/sqrt(64)
    float m = -3.4e38f;
    for (int e = 0; e < 64; ++e) m = fmaxf(m, L[t][e] * scale);
    float s = 0.f;
    for (int e = 0; e < 64; ++e) s += __expf(L[t][e] * scale - m);
    const float inv = 1.0f / s;
    __bf16* dst = attnw + (size_t)bh * 4096 + t * 64;
    for (int e = 0; e < 64; ++e)
      dst[e] = (__bf16)(__expf(L[t][e] * scale - m) * inv);
  }
}

// ----------------------- AV GEMM: out = attn @ v ---------------------------
// Grid: (N/128, B*H). Block 256 thr, output tile 64(d) x 128(n), K=64 (e).
__global__ void av_gemm_kernel(const __bf16* __restrict__ qkvbf,
                               const __bf16* __restrict__ attnw,
                               __bf16* __restrict__ aobf) {
  __shared__ __bf16 attnS[64][64];  // [d][e]  8KB
  __shared__ __bf16 Vst[128][64];   // [n][e] 16KB (transposed)

  const int bh = blockIdx.y;
  const int b  = bh >> 3;
  const int h  = bh & 7;
  const int n0 = blockIdx.x * 128;
  const __bf16* vp = qkvbf + ((size_t)b * ODIM + 2 * CDIM + h * DHEAD) * NDIM;

  const int t    = threadIdx.x;
  const int lane = t & 31;
  const int w    = t >> 5;
  const int lrow = lane & 15;
  const int lhi  = lane >> 4;

  // stage attn 64x64 via async (2 x 16B per thread, linear copy)
#pragma unroll
  for (int cp = 0; cp < 2; ++cp) {
    const int idx = t + cp * 256;
    const int lin = idx * 8;
    async_g2l_b128(&attnS[0][0] + lin, attnw + (size_t)bh * 4096 + lin);
  }
  { // stage v tile 64(e) x 128(n) transposed into Vst[n][e], packed b32 stores
    const int nb = (t & 31) * 4;     // 4 n-values
    const int ep = t >> 5;           // e row-pair base (0..7)
#pragma unroll
    for (int rep = 0; rep < 4; ++rep) {
      const int e0 = (ep + rep * 8) * 2;
      const int e1 = e0 + 1;
      v4bf a = *(const v4bf*)(vp + (size_t)e0 * NDIM + n0 + nb);
      v4bf c = *(const v4bf*)(vp + (size_t)e1 * NDIM + n0 + nb);
#pragma unroll
      for (int i = 0; i < 4; ++i)
        *(unsigned*)(&Vst[nb + i][e0]) = pack_bf16(a[i], c[i]);
    }
  }
  wait_async0();
  __syncthreads();

  v8f acc[4] = {vzero8(), vzero8(), vzero8(), vzero8()};
  const int nl = w * 16 + lrow;
#pragma unroll
  for (int ks = 0; ks < 2; ++ks) {
    v16bf bf = *(const v16bf*)(&Vst[nl][ks * 32 + lhi * 16]);
#pragma unroll
    for (int m2 = 0; m2 < 4; ++m2) {
      const int arow = m2 * 16 + lrow;
      v16bf a = *(const v16bf*)(&attnS[arow][ks * 32 + lhi * 16]);
      acc[m2] = wmma_bf16(a, bf, acc[m2]);
    }
  }

  const int n = n0 + w * 16 + lrow;
#pragma unroll
  for (int m2 = 0; m2 < 4; ++m2) {
#pragma unroll
    for (int i = 0; i < 8; ++i) {
      const int c = h * DHEAD + m2 * 16 + i + 8 * lhi;
      aobf[((size_t)b * CDIM + c) * NDIM + n] = (__bf16)acc[m2][i];
    }
  }
}

// --------------------- GEMM 3: proj + bias + residual ----------------------
// out[b][o][n] = sum_c P[o][c]*AO[b][c][n] + pb[o] + x[b][o][n]   (f32 out)
__global__ void proj_gemm_kernel(const __bf16* __restrict__ Pbf,
                                 const __bf16* __restrict__ AO,
                                 const float*  __restrict__ pb,
                                 const float*  __restrict__ X,
                                 float* __restrict__ Out) {
  __shared__ __bf16 As[2][128][32];
  __shared__ __bf16 Bst[2][64][32];

  const int b  = blockIdx.z;
  const int m0 = blockIdx.y * 128;
  const int n0 = blockIdx.x * 64;
  const __bf16* AOb = AO + (size_t)b * CDIM * NDIM;

  const int t    = threadIdx.x;
  const int lane = t & 31;
  const int w    = t >> 5;
  const int lrow = lane & 15;
  const int lhi  = lane >> 4;
  const int mw   = w & 3;
  const int nw   = w >> 2;

  const int ar0s = (t)       >> 2, ac0s = ((t)       & 3) * 8;
  const int ar1s = (t + 256) >> 2, ac1s = ((t + 256) & 3) * 8;
  const int nb  = (t & 15) * 4;
  const int rk0 = (t >> 4) * 2;
  const int rk1 = rk0 + 1;

  v8f acc00 = vzero8(), acc01 = vzero8(), acc10 = vzero8(), acc11 = vzero8();

  // prologue: stage tile 0
  async_g2l_b128(&As[0][ar0s][ac0s], Pbf + (size_t)(m0 + ar0s) * CDIM + ac0s);
  async_g2l_b128(&As[0][ar1s][ac1s], Pbf + (size_t)(m0 + ar1s) * CDIM + ac1s);
  {
    v4bf g0 = *(const v4bf*)(AOb + (size_t)rk0 * NDIM + n0 + nb);
    v4bf g1 = *(const v4bf*)(AOb + (size_t)rk1 * NDIM + n0 + nb);
#pragma unroll
    for (int i = 0; i < 4; ++i)
      *(unsigned*)(&Bst[0][nb + i][rk0]) = pack_bf16(g0[i], g1[i]);
  }
  wait_async0();
  __syncthreads();

  const int KT = CDIM / 32;
  for (int kt = 0; kt < KT; ++kt) {
    const int cur = kt & 1, nxt = cur ^ 1;
    const bool has_next = (kt + 1) < KT;
    v4bf g0, g1;
    if (has_next) {
      const int k0n = (kt + 1) * 32;
      async_g2l_b128(&As[nxt][ar0s][ac0s],
                     Pbf + (size_t)(m0 + ar0s) * CDIM + k0n + ac0s);
      async_g2l_b128(&As[nxt][ar1s][ac1s],
                     Pbf + (size_t)(m0 + ar1s) * CDIM + k0n + ac1s);
      g0 = *(const v4bf*)(AOb + (size_t)(k0n + rk0) * NDIM + n0 + nb);
      g1 = *(const v4bf*)(AOb + (size_t)(k0n + rk1) * NDIM + n0 + nb);
    }

    const int ar = mw * 32 + lrow;
    v16bf a0 = *(const v16bf*)(&As[cur][ar][lhi * 16]);
    v16bf a1 = *(const v16bf*)(&As[cur][ar + 16][lhi * 16]);
    v16bf b0 = *(const v16bf*)(&Bst[cur][nw * 32 + lrow][lhi * 16]);
    v16bf b1 = *(const v16bf*)(&Bst[cur][nw * 32 + 16 + lrow][lhi * 16]);
    acc00 = wmma_bf16(a0, b0, acc00);
    acc01 = wmma_bf16(a0, b1, acc01);
    acc10 = wmma_bf16(a1, b0, acc10);
    acc11 = wmma_bf16(a1, b1, acc11);

    if (has_next) {
#pragma unroll
      for (int i = 0; i < 4; ++i)
        *(unsigned*)(&Bst[nxt][nb + i][rk0]) = pack_bf16(g0[i], g1[i]);
    }
    wait_async0();
    __syncthreads();
  }

  const int nA = n0 + nw * 32 + lrow;
  const int nB = nA + 16;
#pragma unroll
  for (int i = 0; i < 8; ++i) {
    const int o0 = m0 + mw * 32 + i + 8 * lhi;
    const int o1 = o0 + 16;
    const float bo0 = pb[o0], bo1 = pb[o1];
    const size_t i00 = ((size_t)b * CDIM + o0) * NDIM + nA;
    const size_t i01 = ((size_t)b * CDIM + o0) * NDIM + nB;
    const size_t i10 = ((size_t)b * CDIM + o1) * NDIM + nA;
    const size_t i11 = ((size_t)b * CDIM + o1) * NDIM + nB;
    Out[i00] = acc00[i] + bo0 + X[i00];
    Out[i01] = acc01[i] + bo0 + X[i01];
    Out[i10] = acc10[i] + bo1 + X[i10];
    Out[i11] = acc11[i] + bo1 + X[i11];
  }
}

// ------------------------------- launcher ----------------------------------
extern "C" void kernel_launch(void* const* d_in, const int* in_sizes, int n_in,
                              void* d_out, int out_size, void* d_ws, size_t ws_size,
                              hipStream_t stream) {
  const float* x      = (const float*)d_in[0];
  const float* qkv_w  = (const float*)d_in[1];
  const float* qkv_b  = (const float*)d_in[2];
  const float* proj_w = (const float*)d_in[3];
  const float* proj_b = (const float*)d_in[4];
  float* out = (float*)d_out;

  // workspace carve (all offsets 256B aligned)
  char* p = (char*)d_ws;
  __bf16* wqkv_bf  = (__bf16*)p; p += (size_t)ODIM * CDIM * 2;           // 1.5 MB
  __bf16* wproj_bf = (__bf16*)p; p += (size_t)CDIM * CDIM * 2;           // 0.5 MB
  __bf16* qkv_bf   = (__bf16*)p; p += (size_t)BATCH * ODIM * NDIM * 2;   // 96 MB
  __bf16* ao_bf    = (__bf16*)p; p += (size_t)BATCH * CDIM * NDIM * 2;   // 32 MB
  __bf16* attn_ws  = (__bf16*)p; /* 64 * 64*64 * 2 = 512 KB */

  f32_to_bf16_kernel<<<3072, 256, 0, stream>>>(qkv_w, wqkv_bf, ODIM * CDIM);
  f32_to_bf16_kernel<<<1024, 256, 0, stream>>>(proj_w, wproj_bf, CDIM * CDIM);

  qkv_gemm_kernel<<<dim3(NDIM / 64, ODIM / 128, BATCH), 256, 0, stream>>>(
      wqkv_bf, x, qkv_b, qkv_bf);

  attn_logits_kernel<<<BATCH * HEADS, 256, 0, stream>>>(qkv_bf, attn_ws);

  av_gemm_kernel<<<dim3(NDIM / 128, BATCH * HEADS), 256, 0, stream>>>(
      qkv_bf, attn_ws, ao_bf);

  proj_gemm_kernel<<<dim3(NDIM / 64, CDIM / 128, BATCH), 256, 0, stream>>>(
      wproj_bf, ao_bf, proj_b, x, out);
}
